// mixprop_11338713661508
// MI455X (gfx1250) — matrix operand order repaired and checked by reading, not compiled, not run on hardware
//
#include <hip/hip_runtime.h>

typedef __bf16 bf16_t;
typedef __attribute__((ext_vector_type(16))) __bf16 v16bf;
typedef __attribute__((ext_vector_type(8)))  __bf16 v8bf;
typedef __attribute__((ext_vector_type(4)))  __bf16 v4bf;
typedef __attribute__((ext_vector_type(8)))  float  v8f;

#define NN    4096            // nodes
#define NCOLS 384             // 4 comps * (B=2 * C/4=4 * T=12)
#define TM    128
#define TN    128
#define TK    32

// ---------------- workspace layout (bytes) ----------------
#define OFF_A     ((size_t)0)                         // bf16 [4][NN][NN]   128 MB
#define SZ_A      ((size_t)4 * NN * NN * 2)
#define OFF_SCALE (OFF_A + SZ_A)                      // f32  [3][NN]
#define SZ_SCALE  ((size_t)3 * NN * 4)
#define OFF_HT    (OFF_SCALE + SZ_SCALE)              // bf16 [NCOLS][NN]   3 MB
#define SZ_HT     ((size_t)NCOLS * NN * 2)
#define OFF_XP    (OFF_HT + SZ_HT)                    // f32  [NN][NCOLS]
#define SZ_XP     ((size_t)NN * NCOLS * 4)
#define OFF_P     (OFF_XP + SZ_XP)                    // f32  [4][NN][NCOLS]
#define SZ_P      ((size_t)4 * NN * NCOLS * 4)
#define OFF_H1    (OFF_P + SZ_P)                      // f32  [NN][NCOLS]
#define SZ_H1     ((size_t)NN * NCOLS * 4)
#define OFF_H2    (OFF_H1 + SZ_H1)
#define SZ_H2     ((size_t)NN * NCOLS * 4)
#define WS_NEED   (OFF_H2 + SZ_H2)

// ---------------- 1) row sums of (A+I) for i/j/k ----------------
__global__ __launch_bounds__(256)
void mp_rowsum(const float* __restrict__ Ai, const float* __restrict__ Aj,
               const float* __restrict__ Ak, float* __restrict__ scale) {
  const float* A = (blockIdx.y == 0) ? Ai : (blockIdx.y == 1) ? Aj : Ak;
  const int row = blockIdx.x;
  __shared__ float red[256];
  float s = 0.f;
  for (int w = threadIdx.x; w < NN; w += 256) s += A[(size_t)row * NN + w];
  red[threadIdx.x] = s;
  __syncthreads();
  for (int off = 128; off > 0; off >>= 1) {
    if ((int)threadIdx.x < off) red[threadIdx.x] += red[threadIdx.x + off];
    __syncthreads();
  }
  if (threadIdx.x == 0)
    scale[blockIdx.y * NN + row] = 1.0f / (red[0] + 1.0f);   // (A+I) rowsum
}

// ---------------- 2) normalize + convert to bf16 ----------------
__global__ __launch_bounds__(256)
void mp_convert(const float* __restrict__ Ar, const float* __restrict__ Ai,
                const float* __restrict__ Aj, const float* __restrict__ Ak,
                const float* __restrict__ scale, bf16_t* __restrict__ Abf) {
  const int s = blockIdx.y;
  const float* A = (s == 0) ? Ar : (s == 1) ? Ai : (s == 2) ? Aj : Ak;
  const size_t base = ((size_t)blockIdx.x * 256 + threadIdx.x) * 4;  // element idx in matrix
  const int v = (int)(base >> 12);
  const int w = (int)(base & 4095);
  float4 val = *(const float4*)&A[(size_t)v * NN + w];
  float sc = (s == 0) ? 1.0f : scale[(s - 1) * NN + v];
  float f[4] = {val.x, val.y, val.z, val.w};
  if (s != 0) {
#pragma unroll
    for (int u = 0; u < 4; ++u) {
      if (v == w + u) f[u] += 1.0f;   // self loop
      f[u] *= sc;
    }
  }
  v4bf o = {(bf16_t)f[0], (bf16_t)f[1], (bf16_t)f[2], (bf16_t)f[3]};
  *(v4bf*)&Abf[(size_t)s * NN * NN + base] = o;
}

// ---------------- 3) pack x -> Xp (f32 [n][col]) and Ht (bf16 [col][n]) ----
__global__ __launch_bounds__(256)
void mp_pack(const float* __restrict__ x, float* __restrict__ Xp,
             bf16_t* __restrict__ Ht) {
  const size_t id = (size_t)blockIdx.x * 256 + threadIdx.x;  // NN*NCOLS
  const int n = (int)(id / NCOLS);
  const int col = (int)(id % NCOLS);
  const int comp = col / 96, rem = col % 96;
  const int bc = rem / 12, t = rem % 12;
  const int b = bc >> 2, cq = bc & 3;
  const int ch = comp * 4 + cq;
  const float v = x[(((size_t)b * 16 + ch) * NN + n) * 12 + t];
  Xp[id] = v;
  Ht[(size_t)col * NN + n] = (bf16_t)v;
}

// ---------------- 4) bf16 WMMA GEMM: P[s] = A_s @ H  ----------------
// A: bf16 [NN][NN] row-major (K contiguous); Ht: bf16 [NCOLS][NN] (K contiguous)
// P: f32 [4][NN][NCOLS]
__global__ __launch_bounds__(256)
void mp_gemm(const bf16_t* __restrict__ Aall, const bf16_t* __restrict__ Ht,
             float* __restrict__ P) {
  __shared__ __align__(32) bf16_t As[TM * TK];  // 8 KB
  __shared__ __align__(32) bf16_t Bs[TN * TK];  // 8 KB

  const int s  = blockIdx.z;
  const int rt = blockIdx.y * TM;   // output row (node v) base
  const int ct = blockIdx.x * TN;   // output col base
  const bf16_t* A = Aall + (size_t)s * NN * NN;

  const int tid  = threadIdx.x;
  const int wave = tid >> 5;
  const int lane = tid & 31;
  const int l16  = lane >> 4;       // half-wave select
  const int lm   = lane & 15;

  const int wr = (wave & 3) * 32;   // wave row offset in tile (2 x 16)
  const int wc = (wave >> 2) * 64;  // wave col offset in tile (4 x 16)

  v8f acc[2][4];
#pragma unroll
  for (int i = 0; i < 2; ++i)
#pragma unroll
    for (int j = 0; j < 4; ++j)
      acc[i][j] = (v8f){0.f, 0.f, 0.f, 0.f, 0.f, 0.f, 0.f, 0.f};

  // global->LDS: each thread moves 16 bf16 (32B): row = tid/2, koff = (tid&1)*16
  const int lr = tid >> 1;
  const int lk = (tid & 1) * 16;

  for (int k0 = 0; k0 < NN; k0 += TK) {
    const uint4* gA = (const uint4*)(A + (size_t)(rt + lr) * NN + k0 + lk);
    const uint4 a0 = gA[0], a1 = gA[1];
    const uint4* gB = (const uint4*)(Ht + (size_t)(ct + lr) * NN + k0 + lk);
    const uint4 b0 = gB[0], b1 = gB[1];
    __syncthreads();
    *(uint4*)&As[lr * TK + lk]     = a0;
    *(uint4*)&As[lr * TK + lk + 8] = a1;
    *(uint4*)&Bs[lr * TK + lk]     = b0;
    *(uint4*)&Bs[lr * TK + lk + 8] = b1;
    __syncthreads();

    // A fragments: lane lm = M row; halves at K = l16*8 and 16 + l16*8
    v16bf afrag[2];
#pragma unroll
    for (int i = 0; i < 2; ++i) {
      const v8bf lo = *(const v8bf*)&As[(wr + i * 16 + lm) * TK + l16 * 8];
      const v8bf hi = *(const v8bf*)&As[(wr + i * 16 + lm) * TK + 16 + l16 * 8];
      afrag[i] = __builtin_shufflevector(lo, hi, 0, 1, 2, 3, 4, 5, 6, 7,
                                         8, 9, 10, 11, 12, 13, 14, 15);
    }
    // B fragments: lane lm = N col; halves at K = l16*8 and 16 + l16*8
    v16bf bfrag[4];
#pragma unroll
    for (int j = 0; j < 4; ++j) {
      const v8bf lo = *(const v8bf*)&Bs[(wc + j * 16 + lm) * TK + l16 * 8];
      const v8bf hi = *(const v8bf*)&Bs[(wc + j * 16 + lm) * TK + 16 + l16 * 8];
      bfrag[j] = __builtin_shufflevector(lo, hi, 0, 1, 2, 3, 4, 5, 6, 7,
                                         8, 9, 10, 11, 12, 13, 14, 15);
    }

#pragma unroll
    for (int i = 0; i < 2; ++i)
#pragma unroll
      for (int j = 0; j < 4; ++j)
        acc[i][j] = __builtin_amdgcn_wmma_f32_16x16x32_bf16(
            false, afrag[i], false, bfrag[j], (short)0, acc[i][j],
            false, false);
  }

  // store: C/D layout: lanes 0-15 -> M=p, lanes 16-31 -> M=p+8; N = lm
  float* Pout = P + (size_t)s * NN * NCOLS;
#pragma unroll
  for (int i = 0; i < 2; ++i) {
    const int vbase = rt + wr + i * 16 + (l16 ? 8 : 0);
#pragma unroll
    for (int j = 0; j < 4; ++j) {
      const int col = ct + wc + j * 16 + lm;
#pragma unroll
      for (int p = 0; p < 8; ++p)
        Pout[(size_t)(vbase + p) * NCOLS + col] = acc[i][j][p];
    }
  }
}

// ---------------- 5) quaternion sign-combine + mixprop ----------------
__global__ __launch_bounds__(256)
void mp_combine(const float* __restrict__ P, const float* __restrict__ Xp,
                float* __restrict__ Hout, bf16_t* __restrict__ Ht) {
  const size_t id = (size_t)blockIdx.x * 256 + threadIdx.x;  // NN*NCOLS
  const int v = (int)(id / NCOLS);
  const int col = (int)(id % NCOLS);
  const int c = col / 96, q = col % 96;
  const int   inp[4][4] = {{0, 1, 2, 3}, {1, 0, 3, 2}, {2, 3, 0, 1}, {3, 2, 1, 0}};
  const float sgn[4][4] = {{1, -1, -1, -1}, {1, 1, 1, -1}, {1, -1, 1, 1}, {1, 1, -1, 1}};
  float acc = 0.f;
#pragma unroll
  for (int s = 0; s < 4; ++s)
    acc += sgn[c][s] * P[((size_t)s * NN + v) * NCOLS + inp[c][s] * 96 + q];
  const float h = 0.05f * Xp[id] + 0.95f * acc;
  Hout[id] = h;
  Ht[(size_t)col * NN + v] = (bf16_t)h;
}

// ---------------- 6) final hamilton einsum ----------------
__global__ __launch_bounds__(256)
void mp_out(const float* __restrict__ Xp, const float* __restrict__ H1,
            const float* __restrict__ H2, const float* __restrict__ weight,
            float* __restrict__ out) {
  __shared__ float ham[3][16][16];
  for (int idx = threadIdx.x; idx < 768; idx += 256) {
    const int k = idx >> 8, rem = idx & 255;
    const int ci = rem >> 4, co = rem & 15;
    const int a = ci >> 2, p = ci & 3, bb = co >> 2, q = co & 3;
    const int   cmp[4][4] = {{0, 1, 2, 3}, {1, 0, 3, 2}, {2, 3, 0, 1}, {3, 2, 1, 0}};
    const float sg[4][4]  = {{1, -1, -1, -1}, {1, 1, -1, 1}, {1, 1, 1, -1}, {1, -1, 1, 1}};
    ham[k][ci][co] = sg[bb][a] * weight[k * 64 + p * 16 + cmp[bb][a] * 4 + q];
  }
  __syncthreads();
  const size_t id = (size_t)blockIdx.x * 256 + threadIdx.x;  // 2*NN*12
  const int t = (int)(id % 12);
  const int n = (int)((id / 12) & (NN - 1));
  const int b = (int)(id / ((size_t)12 * NN));
  float acc[16];
#pragma unroll
  for (int j = 0; j < 16; ++j) acc[j] = 0.f;
  const float* bufs[3] = {Xp, H1, H2};
  for (int k = 0; k < 3; ++k) {
    const float* buf = bufs[k];
#pragma unroll
    for (int i = 0; i < 16; ++i) {
      const int comp = i >> 2, cq = i & 3;
      const float v = buf[(size_t)n * NCOLS + comp * 96 + (b * 4 + cq) * 12 + t];
#pragma unroll
      for (int j = 0; j < 16; ++j) acc[j] += v * ham[k][i][j];
    }
  }
#pragma unroll
  for (int j = 0; j < 16; ++j)
    out[(((size_t)b * 16 + j) * NN + n) * 12 + t] = acc[j];
}

extern "C" void kernel_launch(void* const* d_in, const int* in_sizes, int n_in,
                              void* d_out, int out_size, void* d_ws, size_t ws_size,
                              hipStream_t stream) {
  const float* x  = (const float*)d_in[0];
  const float* ar = (const float*)d_in[1];
  const float* ai = (const float*)d_in[2];
  const float* aj = (const float*)d_in[3];
  const float* ak = (const float*)d_in[4];
  const float* w  = (const float*)d_in[5];
  float* out = (float*)d_out;

  if (ws_size < WS_NEED) return;  // insufficient scratch; avoid OOB

  char* ws = (char*)d_ws;
  bf16_t* Abf   = (bf16_t*)(ws + OFF_A);
  float*  scale = (float*)(ws + OFF_SCALE);
  bf16_t* Ht    = (bf16_t*)(ws + OFF_HT);
  float*  Xp    = (float*)(ws + OFF_XP);
  float*  P     = (float*)(ws + OFF_P);
  float*  H1    = (float*)(ws + OFF_H1);
  float*  H2    = (float*)(ws + OFF_H2);

  // 1) row sums for i/j/k
  mp_rowsum<<<dim3(NN, 3), 256, 0, stream>>>(ai, aj, ak, scale);
  // 2) normalize + bf16 convert all four adjacencies
  mp_convert<<<dim3((NN * NN) / (4 * 256), 4), 256, 0, stream>>>(ar, ai, aj, ak, scale, Abf);
  // 3) pack x
  mp_pack<<<(NN * NCOLS) / 256, 256, 0, stream>>>(x, Xp, Ht);

  const dim3 ggrid(NCOLS / TN, NN / TM, 4);
  // 4-5) propagation step 1
  mp_gemm<<<ggrid, 256, 0, stream>>>(Abf, Ht, P);
  mp_combine<<<(NN * NCOLS) / 256, 256, 0, stream>>>(P, Xp, H1, Ht);
  // propagation step 2 (A now resident in 192MB L2)
  mp_gemm<<<ggrid, 256, 0, stream>>>(Abf, Ht, P);
  mp_combine<<<(NN * NCOLS) / 256, 256, 0, stream>>>(P, Xp, H2, Ht);
  // 6) output einsum
  mp_out<<<(2 * NN * 12) / 256, 256, 0, stream>>>(Xp, H1, H2, w, out);
}
